// AttentionBlock_53102975647880
// MI455X (gfx1250) — matrix operand order repaired and checked by reading, not compiled, not run on hardware
//
#include <hip/hip_runtime.h>
#include <stdint.h>

typedef __attribute__((ext_vector_type(16))) _Float16 v16h;
typedef __attribute__((ext_vector_type(8)))  _Float16 v8h;
typedef __attribute__((ext_vector_type(8)))  float    v8f;
typedef __attribute__((ext_vector_type(4)))  float    v4f;

constexpr int Bsz  = 4;
constexpr int Nseq = 2048;
constexpr int Dm   = 768;
constexpr int Hh   = 12;
constexpr int HD   = 64;

// ---------------------------------------------------------------------------
// WMMA helper: D = A(16x32 f16) * B(32x16 f16) + C(16x16 f32)
// ---------------------------------------------------------------------------
__device__ __forceinline__ v8f wmma16(v16h a, v16h b, v8f c) {
  return __builtin_amdgcn_wmma_f32_16x16x32_f16(
      /*neg_a=*/false, a, /*neg_b=*/false, b,
      /*c_mod=*/(short)0, c, /*reuse_a=*/false, /*reuse_b=*/false);
}

// Async global -> LDS copy, 16 bytes/lane, tracked by ASYNCcnt.
__device__ __forceinline__ void async_ld_b128(uint32_t lds_off, const void* g) {
  asm volatile("global_load_async_to_lds_b128 %0, %1, off"
               :: "v"(lds_off), "v"(g) : "memory");
}
__device__ __forceinline__ uint32_t lds_addr(const void* p) {
  return (uint32_t)(uintptr_t)p;   // LDS aperture: low 32 bits of the pointer are the LDS address
}

// A-matrix (16x32, f16): row M = lane&15, element e -> K = (e/8)*16+(lane>>4)*8+e%8
__device__ __forceinline__ v16h ldA_f16(const _Float16* src, int stride, int lane) {
  const int row = lane & 15, hf = (lane >> 4) & 1;
  const _Float16* p0 = src + (size_t)row * stride + hf * 8;
  v8h lo = *(const v8h*)p0;
  v8h hi = *(const v8h*)(p0 + 16);
  v16h a;
#pragma unroll
  for (int i = 0; i < 8; ++i) { a[i] = lo[i]; a[8 + i] = hi[i]; }
  return a;
}

// B-matrix (32x16, f16): col N = lane&15, element e -> K = (e/8)*16+(lane>>4)*8+e%8.
// `bt` is n-major (Bt[n][k], row stride strideK) -> two runs of 8 contiguous halfs/lane, 16 apart.
__device__ __forceinline__ v16h ldB_f16(const _Float16* bt, int strideK, int lane) {
  const int n = lane & 15, hf = (lane >> 4) & 1;
  const _Float16* p = bt + (size_t)n * strideK + hf * 8;
  v8h lo = *(const v8h*)p;
  v8h hi = *(const v8h*)(p + 16);
  v16h r;
#pragma unroll
  for (int i = 0; i < 8; ++i) { r[i] = lo[i]; r[8 + i] = hi[i]; }
  return r;
}

// ---------------------------------------------------------------------------
// Kernel 0a: LDS-tiled transpose + f32->f16 convert.
// src: rows x cols (row-major f32) -> dst: cols x rows (row-major f16)
// rows, cols multiples of 32. 256 threads = 32x8.
// ---------------------------------------------------------------------------
__global__ void __launch_bounds__(256)
k_transpose_f16(const float* __restrict__ src, _Float16* __restrict__ dst,
                int rows, int cols) {
  __shared__ float tile[32][33];
  const int c0 = blockIdx.x * 32, r0 = blockIdx.y * 32;
  const int tx = threadIdx.x & 31, ty = threadIdx.x >> 5;
#pragma unroll
  for (int i = 0; i < 32; i += 8)
    tile[ty + i][tx] = src[(size_t)(r0 + ty + i) * cols + c0 + tx];
  __syncthreads();
#pragma unroll
  for (int i = 0; i < 32; i += 8)
    dst[(size_t)(c0 + ty + i) * rows + r0 + tx] = (_Float16)tile[tx][ty + i];
}

// ---------------------------------------------------------------------------
// Kernel 0b: elementwise f32 -> f16 convert (for x), fully coalesced
// ---------------------------------------------------------------------------
__global__ void __launch_bounds__(256)
k_cvt_f16(const float* __restrict__ src, _Float16* __restrict__ dst, int n) {
  for (int i = blockIdx.x * blockDim.x + threadIdx.x; i < n;
       i += gridDim.x * blockDim.x)
    dst[i] = (_Float16)src[i];
}

// ---------------------------------------------------------------------------
// Kernel 1: QKV projection GEMM. Block = 128 rows x 64 cols.
// Double-buffered async LDS B tile of 64n x 64k f16 (8KB); 8 WMMAs / barrier.
// ---------------------------------------------------------------------------
__global__ void __launch_bounds__(256)
k_qkv(const _Float16* __restrict__ xh, const _Float16* __restrict__ Wt,
      const float* __restrict__ bqkv, _Float16* __restrict__ Qh,
      _Float16* __restrict__ Kh, _Float16* __restrict__ Vt) {
  __shared__ _Float16 ldsB[2][64 * 64];
  const int lane = threadIdx.x & 31;
  const int wv   = threadIdx.x >> 5;
  const int NCG  = (3 * Dm) / 64;                 // 36 column groups
  const int nBase   = (blockIdx.x % NCG) * 64;
  const int rowBase = (blockIdx.x / NCG) * 128 + wv * 16;

  // async stage: thread t -> n = t/4, chunks t%4 and t%4+4 (2 x 16B per thread)
  const int tn = threadIdx.x >> 2, tc = threadIdx.x & 3;
  const _Float16* gB = Wt + (size_t)(nBase + tn) * Dm;
  auto issueB = [&](int ks, int bbuf) {
    async_ld_b128(lds_addr(&ldsB[bbuf][tn * 64 + tc * 8]),       gB + ks + tc * 8);
    async_ld_b128(lds_addr(&ldsB[bbuf][tn * 64 + (tc + 4) * 8]), gB + ks + (tc + 4) * 8);
  };

  v8f acc[4];
#pragma unroll
  for (int t = 0; t < 4; ++t)
#pragma unroll
    for (int v = 0; v < 8; ++v) acc[t][v] = 0.0f;

  issueB(0, 0);
  for (int ks = 0; ks < Dm; ks += 64) {
    const int bb = (ks >> 6) & 1;
    // A tiles (global f16, independent of LDS) issued before the wait/barrier
    v16h a0 = ldA_f16(xh + (size_t)rowBase * Dm + ks, Dm, lane);
    v16h a1 = ldA_f16(xh + (size_t)rowBase * Dm + ks + 32, Dm, lane);
    if (ks + 64 < Dm) {
      issueB(ks + 64, bb ^ 1);
      asm volatile("s_wait_asynccnt 0x2" ::: "memory");
    } else {
      asm volatile("s_wait_asynccnt 0x0" ::: "memory");
    }
    __syncthreads();

    // batch the DS loads ahead of the WMMA group
    v16h b0l = ldB_f16(&ldsB[bb][0 * 16 * 64], 64, lane);
    v16h b0h = ldB_f16(&ldsB[bb][0 * 16 * 64 + 32], 64, lane);
    v16h b1l = ldB_f16(&ldsB[bb][1 * 16 * 64], 64, lane);
    v16h b1h = ldB_f16(&ldsB[bb][1 * 16 * 64 + 32], 64, lane);
    v16h b2l = ldB_f16(&ldsB[bb][2 * 16 * 64], 64, lane);
    v16h b2h = ldB_f16(&ldsB[bb][2 * 16 * 64 + 32], 64, lane);
    v16h b3l = ldB_f16(&ldsB[bb][3 * 16 * 64], 64, lane);
    v16h b3h = ldB_f16(&ldsB[bb][3 * 16 * 64 + 32], 64, lane);
    acc[0] = wmma16(a0, b0l, acc[0]);  acc[0] = wmma16(a1, b0h, acc[0]);
    acc[1] = wmma16(a0, b1l, acc[1]);  acc[1] = wmma16(a1, b1h, acc[1]);
    acc[2] = wmma16(a0, b2l, acc[2]);  acc[2] = wmma16(a1, b2h, acc[2]);
    acc[3] = wmma16(a0, b3l, acc[3]);  acc[3] = wmma16(a1, b3h, acc[3]);

    asm volatile("s_wait_dscnt 0x0" ::: "memory");
    __syncthreads();                              // before buffer reuse
  }

  const int which = nBase / Dm;                   // 0=Q 1=K 2=V
  const int h     = (nBase % Dm) / HD;
  const int hf = (lane >> 4) & 1, cn = lane & 15;
#pragma unroll
  for (int t = 0; t < 4; ++t) {
    const int d  = t * 16 + cn;
    const float bb = bqkv[nBase + t * 16 + cn];
#pragma unroll
    for (int v = 0; v < 8; ++v) {
      const int rowg = rowBase + v + hf * 8;
      const int b  = rowg >> 11;
      const int nn = rowg & (Nseq - 1);
      const float val = acc[t][v] + bb;
      const size_t head = (size_t)b * Hh + h;
      if (which == 0)
        Qh[(head * Nseq + nn) * HD + d] = (_Float16)(val * 0.125f);  // 1/sqrt(64)
      else if (which == 1)
        Kh[(head * Nseq + nn) * HD + d] = (_Float16)val;
      else
        Vt[(head * HD + d) * Nseq + nn] = (_Float16)val;             // transposed
    }
  }
}

// ---------------------------------------------------------------------------
// Kernel 2: flash attention, 64-key blocks. Block = 128 query rows of one
// (b,h); async double-buffered K (64key x 64d) and V^T (64d x 64key) tiles.
// blockIdx = (h * 16 + qgroup) * B + b  (b fastest -> bias reuse in L2)
// ---------------------------------------------------------------------------
__global__ void __launch_bounds__(256)
k_attn(const _Float16* __restrict__ Qh, const _Float16* __restrict__ Kh,
       const _Float16* __restrict__ Vt, const float* __restrict__ bias,
       _Float16* __restrict__ aout) {
  __shared__ _Float16 ldsK[2][64 * 64];   // [key][d]   8KB each
  __shared__ _Float16 ldsV[2][64 * 64];   // [d][key]   8KB each
  __shared__ _Float16 ldsP[8][16 * 64];   // per-wave P tile (C->A bridge)

  const int lane = threadIdx.x & 31;
  const int wv   = threadIdx.x >> 5;
  const int b    = blockIdx.x % Bsz;
  const int tmp  = blockIdx.x / Bsz;
  const int qg   = tmp % (Nseq / 128);
  const int h    = tmp / (Nseq / 128);
  const int qBase = qg * 128 + wv * 16;

  const size_t head = (size_t)b * Hh + h;
  const _Float16* Qp = Qh + (head * Nseq + qBase) * HD;
  const _Float16* Kp = Kh + head * Nseq * HD;
  const _Float16* Vp = Vt + head * HD * Nseq;
  const float* biasp = bias + ((size_t)h * Nseq + qBase) * Nseq;

  const int hf = (lane >> 4) & 1, cn = lane & 15;

  const v16h aq0 = ldA_f16(Qp, HD, lane);        // d = 0..31  (Q pre-scaled)
  const v16h aq1 = ldA_f16(Qp + 32, HD, lane);   // d = 32..63

  // async stage: thread t -> row = t/4 (key for K, d for V), chunks t%4, t%4+4
  const int tr = threadIdx.x >> 2, tc = threadIdx.x & 3;
  auto issueKV = [&](int j, int bbuf) {
    const _Float16* gk = Kp + (size_t)(j + tr) * HD;
    const _Float16* gv = Vp + (size_t)tr * Nseq + j;
    async_ld_b128(lds_addr(&ldsK[bbuf][tr * 64 + tc * 8]),       gk + tc * 8);
    async_ld_b128(lds_addr(&ldsK[bbuf][tr * 64 + (tc + 4) * 8]), gk + (tc + 4) * 8);
    async_ld_b128(lds_addr(&ldsV[bbuf][tr * 64 + tc * 8]),       gv + tc * 8);
    async_ld_b128(lds_addr(&ldsV[bbuf][tr * 64 + (tc + 4) * 8]), gv + (tc + 4) * 8);
  };

  v8f O[4];
  float m[8], l[8];
#pragma unroll
  for (int t = 0; t < 4; ++t)
#pragma unroll
    for (int v = 0; v < 8; ++v) O[t][v] = 0.0f;
#pragma unroll
  for (int v = 0; v < 8; ++v) { m[v] = -1e30f; l[v] = 0.0f; }

  _Float16* myP = ldsP[wv];

  issueKV(0, 0);
  for (int j = 0; j < Nseq; j += 64) {
    const int bb = (j >> 6) & 1;
    // bias prefetch (global) before the waits: hidden behind WMMA work
    float bs[4][8];
#pragma unroll
    for (int v = 0; v < 8; ++v) {
      const float* bp = biasp + (size_t)(v + hf * 8) * Nseq + j + cn;
#pragma unroll
      for (int tt = 0; tt < 4; ++tt) bs[tt][v] = bp[tt * 16];
    }
    if (j + 64 < Nseq) {
      issueKV(j + 64, bb ^ 1);
      asm volatile("s_wait_asynccnt 0x4" ::: "memory");  // current 4 done
    } else {
      asm volatile("s_wait_asynccnt 0x0" ::: "memory");
    }
    __syncthreads();

    // S = Q @ K^T for 64 keys (four 16x16 tiles), two tiles per DS batch
    v8f s[4];
#pragma unroll
    for (int tt = 0; tt < 4; ++tt)
#pragma unroll
      for (int v = 0; v < 8; ++v) s[tt][v] = 0.0f;
#pragma unroll
    for (int pp = 0; pp < 4; pp += 2) {
      v16h k0l = ldB_f16(&ldsK[bb][(pp)     * 16 * 64], 64, lane);
      v16h k0h = ldB_f16(&ldsK[bb][(pp)     * 16 * 64 + 32], 64, lane);
      v16h k1l = ldB_f16(&ldsK[bb][(pp + 1) * 16 * 64], 64, lane);
      v16h k1h = ldB_f16(&ldsK[bb][(pp + 1) * 16 * 64 + 32], 64, lane);
      s[pp]     = wmma16(aq0, k0l, s[pp]);
      s[pp]     = wmma16(aq1, k0h, s[pp]);
      s[pp + 1] = wmma16(aq0, k1l, s[pp + 1]);
      s[pp + 1] = wmma16(aq1, k1h, s[pp + 1]);
    }
#pragma unroll
    for (int tt = 0; tt < 4; ++tt)
#pragma unroll
      for (int v = 0; v < 8; ++v) s[tt][v] += bs[tt][v];

    // online softmax; exp results written back into s[][] (P values)
#pragma unroll
    for (int v = 0; v < 8; ++v) {
      float mx = fmaxf(fmaxf(s[0][v], s[1][v]), fmaxf(s[2][v], s[3][v]));
#pragma unroll
      for (int msk = 1; msk < 16; msk <<= 1)
        mx = fmaxf(mx, __shfl_xor(mx, msk, 32));
      const float mnew  = fmaxf(m[v], mx);
      const float alpha = __expf(m[v] - mnew);
      m[v] = mnew;
      float rs = 0.0f;
#pragma unroll
      for (int tt = 0; tt < 4; ++tt) {
        const float e = __expf(s[tt][v] - mnew);
        s[tt][v] = e;
        rs += e;
      }
#pragma unroll
      for (int msk = 1; msk < 16; msk <<= 1)
        rs += __shfl_xor(rs, msk, 32);
      l[v] = l[v] * alpha + rs;
#pragma unroll
      for (int t = 0; t < 4; ++t) O[t][v] *= alpha;
    }
    // P (C layout) -> LDS -> reload in A layout (16 x 64 keys)
#pragma unroll
    for (int v = 0; v < 8; ++v) {
      const int row = v + hf * 8;
#pragma unroll
      for (int tt = 0; tt < 4; ++tt)
        myP[row * 64 + tt * 16 + cn] = (_Float16)s[tt][v];
    }
    asm volatile("s_wait_dscnt 0x0" ::: "memory");
    const v16h ap0 = ldA_f16(myP, 64, lane);        // keys 0..31
    const v16h ap1 = ldA_f16(myP + 32, 64, lane);   // keys 32..63
    // O += P @ V
#pragma unroll
    for (int t = 0; t < 4; ++t) {
      v16h bvl = ldB_f16(&ldsV[bb][t * 16 * 64], 64, lane);       // keys 0..31
      v16h bvh = ldB_f16(&ldsV[bb][t * 16 * 64 + 32], 64, lane);  // keys 32..63
      O[t] = wmma16(ap0, bvl, O[t]);
      O[t] = wmma16(ap1, bvh, O[t]);
    }
    asm volatile("s_wait_dscnt 0x0" ::: "memory");
    __syncthreads();                              // before K/V buffer reuse
  }

  // normalize and write f16 attention output (row-major (B*N, D))
#pragma unroll
  for (int v = 0; v < 8; ++v) {
    const float inv = 1.0f / l[v];
    const size_t rowg = (size_t)b * Nseq + qBase + v + hf * 8;
#pragma unroll
    for (int t = 0; t < 4; ++t)
      aout[rowg * Dm + h * HD + t * 16 + cn] = (_Float16)(O[t][v] * inv);
  }
}

// ---------------------------------------------------------------------------
// Kernel 3: output projection GEMM -> fp32 out. Same staging as k_qkv.
// ---------------------------------------------------------------------------
__global__ void __launch_bounds__(256)
k_proj(const _Float16* __restrict__ ain, const _Float16* __restrict__ Wt,
       const float* __restrict__ bproj, float* __restrict__ out) {
  __shared__ _Float16 ldsB[2][64 * 64];
  const int lane = threadIdx.x & 31;
  const int wv   = threadIdx.x >> 5;
  const int NCG  = Dm / 64;                       // 12
  const int nBase   = (blockIdx.x % NCG) * 64;
  const int rowBase = (blockIdx.x / NCG) * 128 + wv * 16;

  const int tn = threadIdx.x >> 2, tc = threadIdx.x & 3;
  const _Float16* gB = Wt + (size_t)(nBase + tn) * Dm;
  auto issueB = [&](int ks, int bbuf) {
    async_ld_b128(lds_addr(&ldsB[bbuf][tn * 64 + tc * 8]),       gB + ks + tc * 8);
    async_ld_b128(lds_addr(&ldsB[bbuf][tn * 64 + (tc + 4) * 8]), gB + ks + (tc + 4) * 8);
  };

  v8f acc[4];
#pragma unroll
  for (int t = 0; t < 4; ++t)
#pragma unroll
    for (int v = 0; v < 8; ++v) acc[t][v] = 0.0f;

  issueB(0, 0);
  for (int ks = 0; ks < Dm; ks += 64) {
    const int bb = (ks >> 6) & 1;
    v16h a0 = ldA_f16(ain + (size_t)rowBase * Dm + ks, Dm, lane);
    v16h a1 = ldA_f16(ain + (size_t)rowBase * Dm + ks + 32, Dm, lane);
    if (ks + 64 < Dm) {
      issueB(ks + 64, bb ^ 1);
      asm volatile("s_wait_asynccnt 0x2" ::: "memory");
    } else {
      asm volatile("s_wait_asynccnt 0x0" ::: "memory");
    }
    __syncthreads();

    v16h b0l = ldB_f16(&ldsB[bb][0 * 16 * 64], 64, lane);
    v16h b0h = ldB_f16(&ldsB[bb][0 * 16 * 64 + 32], 64, lane);
    v16h b1l = ldB_f16(&ldsB[bb][1 * 16 * 64], 64, lane);
    v16h b1h = ldB_f16(&ldsB[bb][1 * 16 * 64 + 32], 64, lane);
    v16h b2l = ldB_f16(&ldsB[bb][2 * 16 * 64], 64, lane);
    v16h b2h = ldB_f16(&ldsB[bb][2 * 16 * 64 + 32], 64, lane);
    v16h b3l = ldB_f16(&ldsB[bb][3 * 16 * 64], 64, lane);
    v16h b3h = ldB_f16(&ldsB[bb][3 * 16 * 64 + 32], 64, lane);
    acc[0] = wmma16(a0, b0l, acc[0]);  acc[0] = wmma16(a1, b0h, acc[0]);
    acc[1] = wmma16(a0, b1l, acc[1]);  acc[1] = wmma16(a1, b1h, acc[1]);
    acc[2] = wmma16(a0, b2l, acc[2]);  acc[2] = wmma16(a1, b2h, acc[2]);
    acc[3] = wmma16(a0, b3l, acc[3]);  acc[3] = wmma16(a1, b3h, acc[3]);

    asm volatile("s_wait_dscnt 0x0" ::: "memory");
    __syncthreads();
  }

  const int hf = (lane >> 4) & 1, cn = lane & 15;
#pragma unroll
  for (int t = 0; t < 4; ++t) {
    const int col = nBase + t * 16 + cn;
    const float bb = bproj[col];
#pragma unroll
    for (int v = 0; v < 8; ++v) {
      const int rowg = rowBase + v + hf * 8;
      out[(size_t)rowg * Dm + col] = acc[t][v] + bb;
    }
  }
}

// ---------------------------------------------------------------------------
extern "C" void kernel_launch(void* const* d_in, const int* in_sizes, int n_in,
                              void* d_out, int out_size, void* d_ws, size_t ws_size,
                              hipStream_t stream) {
  const float* x     = (const float*)d_in[0];
  const float* bias  = (const float*)d_in[1];
  const float* Wqkv  = (const float*)d_in[2];
  const float* bqkv  = (const float*)d_in[3];
  const float* Wproj = (const float*)d_in[4];
  const float* bproj = (const float*)d_in[5];
  float* out = (float*)d_out;

  char* ws = (char*)d_ws;
  size_t off = 0;
  auto carve = [&](size_t bytes) -> char* {
    char* p = ws + off;
    off += (bytes + 255) & ~(size_t)255;
    return p;
  };
  _Float16* WtQkv  = (_Float16*)carve((size_t)3 * Dm * Dm * 2);          // 3.5 MB
  _Float16* WtProj = (_Float16*)carve((size_t)Dm * Dm * 2);              // 1.2 MB
  _Float16* Qh     = (_Float16*)carve((size_t)Bsz * Hh * Nseq * HD * 2); // 12.6 MB
  _Float16* Kh     = (_Float16*)carve((size_t)Bsz * Hh * Nseq * HD * 2); // 12.6 MB
  _Float16* Vt     = (_Float16*)carve((size_t)Bsz * Hh * Nseq * HD * 2); // 12.6 MB
  _Float16* Aout   = (_Float16*)carve((size_t)Bsz * Nseq * Dm * 2);      // 12.6 MB
  _Float16* Xh     = (_Float16*)carve((size_t)Bsz * Nseq * Dm * 2);      // 12.6 MB

  // prep: x -> f16 (coalesced); weights -> f16 n-major (LDS-tiled transpose)
  k_cvt_f16<<<2048, 256, 0, stream>>>(x, Xh, Bsz * Nseq * Dm);
  k_transpose_f16<<<dim3(3 * Dm / 32, Dm / 32), 256, 0, stream>>>(
      Wqkv, WtQkv, Dm, 3 * Dm);
  k_transpose_f16<<<dim3(Dm / 32, Dm / 32), 256, 0, stream>>>(
      Wproj, WtProj, Dm, Dm);

  // 64 row blocks (128 rows) * 36 col groups
  k_qkv<<<2304, 256, 0, stream>>>(Xh, WtQkv, bqkv, Qh, Kh, Vt);

  // (h * 16 qgroups) * 4 batches = 768 blocks
  k_attn<<<768, 256, 0, stream>>>(Qh, Kh, Vt, bias, Aout);

  // 64 row blocks * 12 col groups
  k_proj<<<768, 256, 0, stream>>>(Aout, WtProj, bproj, out);
}
